// SketchGenerator_69793218560121
// MI455X (gfx1250) — matrix operand order repaired and checked by reading, not compiled, not run on hardware
//
#include <hip/hip_runtime.h>

typedef __attribute__((ext_vector_type(16))) __bf16    v16bf;
typedef __attribute__((ext_vector_type(8)))  float     v8f;
typedef __attribute__((ext_vector_type(4)))  float     f32x4;
typedef __attribute__((ext_vector_type(4)))  unsigned  u32x4;
typedef __attribute__((ext_vector_type(2)))  unsigned  u32x2;

#define N_ROWS 500000
#define DIM    128
#define SCOLS  256      // proj cols
#define HCOLS  512      // hidden cols = 2*S
#define BSEG   64

// ---- workspace layout (bytes) ----
#define WS_W1T    0          // transposed W1 bf16: [512 n][128 k]  = 131072 B
#define WS_W2T    131072     // transposed W2 bf16: [256 n][512 k]  = 262144 B
#define WS_SUMS   393216     // 64*256 f32
#define WS_SUMSQ  458752
#define WS_MAXB   524288     // 64*256 u32 (order-preserving map)
#define WS_MINB   589824

// ---- dynamic LDS layout (bytes) ----
#define SXU_OFF       0          // 128 rows * 68 u32 (bf16 pairs of x); 272B row: 16B aligned, bank phase 4
#define SXU_STRIDE    68         // u32 per row
#define SH_OFF        34816      // 128 rows * 1072 B: bf16[536] for h, reused as f32[256] for proj
#define SH_ROW_BYTES  1072       // 16B aligned, bank phase 12 (conflict-free for 16-lane row stride)
#define SBATCH_OFF    172032     // 128 ints
#define LDS_BYTES     172544

__device__ __forceinline__ unsigned fmap(float f) {
    int i = __float_as_int(f);
    return (unsigned)(i ^ ((i >> 31) | 0x80000000));
}
__device__ __forceinline__ float funmap(unsigned u) {
    int i = (u & 0x80000000u) ? (int)(u ^ 0x80000000u) : ~(int)u;
    return __int_as_float(i);
}

// ------------------------------------------------------------------
// Kernel 0: fp32 weights -> TRANSPOSED bf16 (col-major, so WMMA
// B-tiles are 2x contiguous 16B chunks), re-init stat buffers
// ------------------------------------------------------------------
__global__ void prep_kernel(const float* __restrict__ W1, const float* __restrict__ W2,
                            __bf16* __restrict__ w1t, __bf16* __restrict__ w2t,
                            float* __restrict__ sums, float* __restrict__ sumsq,
                            unsigned* __restrict__ maxb, unsigned* __restrict__ minb) {
    int i = blockIdx.x * blockDim.x + threadIdx.x;
    int stride = gridDim.x * blockDim.x;
    for (int t = i; t < HCOLS * SCOLS; t += stride) {     // 131072
        if (t < DIM * HCOLS) {                            // W1[k][n] -> w1t[n][k]
            int k = t >> 9, n = t & 511;
            w1t[n * DIM + k] = (__bf16)W1[t];
        }
        {                                                 // W2[k][n] -> w2t[n][k]
            int k = t >> 8, n = t & 255;
            w2t[n * HCOLS + k] = (__bf16)W2[t];
        }
        if (t < BSEG * SCOLS) {                           // 16384
            sums[t]  = 0.f;
            sumsq[t] = 0.f;
            maxb[t]  = 0u;
            minb[t]  = 0xFFFFFFFFu;
        }
    }
}

// B-matrix tile loader: lane=col (l16), element e -> k = kbase + 16*half + e
__device__ __forceinline__ v16bf load_btile(const __bf16* __restrict__ wt,
                                            int col, int ldk, int kbase, int half) {
    union { v16bf v; u32x4 q[2]; } t;
    const u32x4* p = (const u32x4*)(wt + col * ldk + kbase + 16 * half);
    t.q[0] = p[0];
    t.q[1] = p[1];
    return t.v;
}

// ------------------------------------------------------------------
// Kernel 1: fused MLP (bf16 WMMA, f32 accum) + segmented partial stats
//   block = 512 threads = 16 wave32, 128 rows per block
//   wave w: h-cols [32w,32w+32), proj-cols [16w,16w+16)
//   B1 (64 VGPR) and B2 (128 VGPR) never live simultaneously
// ------------------------------------------------------------------
__global__ __launch_bounds__(512, 1) void mlp_stats_kernel(
    const float* __restrict__ x, const int* __restrict__ batch,
    const __bf16* __restrict__ w1t, const float* __restrict__ b1,
    const __bf16* __restrict__ w2t, const float* __restrict__ b2,
    float* __restrict__ sums, float* __restrict__ sumsq,
    unsigned* __restrict__ maxb, unsigned* __restrict__ minb)
{
    extern __shared__ __align__(16) unsigned char smem[];
    unsigned* sXu  = (unsigned*)(smem + SXU_OFF);
    int*      sBat = (int*)(smem + SBATCH_OFF);

    const int tid  = threadIdx.x;
    const int w    = tid >> 5;      // wave id 0..15
    const int lane = tid & 31;
    const int half = lane >> 4;
    const int l16  = lane & 15;
    const int r0   = blockIdx.x * 128;

    // ---- stage x row-tile into LDS as packed bf16 (vectorized) ----
    for (int t = tid; t < 128 * 32; t += 512) {
        int row = t >> 5, kq = t & 31;          // kq: group of 4 floats
        int rg = r0 + row;
        f32x4 xa = (f32x4){0.f, 0.f, 0.f, 0.f};
        if (rg < N_ROWS) xa = *(const f32x4*)(x + (long)rg * DIM + kq * 4);
        union { __bf16 h[4]; u32x2 u; } p;
        p.h[0] = (__bf16)xa.x; p.h[1] = (__bf16)xa.y;
        p.h[2] = (__bf16)xa.z; p.h[3] = (__bf16)xa.w;
        *(u32x2*)(sXu + row * SXU_STRIDE + kq * 2) = p.u;
    }
    if (tid < 128) {
        int rg = r0 + tid;
        sBat[tid] = (rg < N_ROWS) ? batch[rg] : -1;
    }

    // ---- B1 weight tiles: wave's 32 h-cols, 4 k-tiles x 2 n-tiles ----
    v16bf B1[4][2];
#pragma unroll
    for (int k1 = 0; k1 < 4; ++k1)
#pragma unroll
        for (int nt = 0; nt < 2; ++nt)
            B1[k1][nt] = load_btile(w1t, w * 32 + nt * 16 + l16, DIM, k1 * 32, half);

    float bias1[2];
#pragma unroll
    for (int nt = 0; nt < 2; ++nt) bias1[nt] = b1[w * 32 + nt * 16 + l16];

    __syncthreads();

    // ---- GEMM1: h = relu(x @ W1 + b1) -> sH (bf16) ----
#pragma unroll 1
    for (int mt = 0; mt < 8; ++mt) {
        v8f acc[2];
#pragma unroll
        for (int nt = 0; nt < 2; ++nt) {
            float bv = bias1[nt];
            acc[nt] = (v8f){bv, bv, bv, bv, bv, bv, bv, bv};
        }
        const int rowu = (mt * 16 + l16) * SXU_STRIDE;
#pragma unroll
        for (int k1 = 0; k1 < 4; ++k1) {
            union { v16bf v; u32x4 q[2]; } A;   // elem e -> k = 32k1 + 16half + e
            A.q[0] = *(const u32x4*)(sXu + rowu + k1 * 16 + 8 * half);
            A.q[1] = *(const u32x4*)(sXu + rowu + k1 * 16 + 4 + 8 * half);
#pragma unroll
            for (int nt = 0; nt < 2; ++nt)
                acc[nt] = __builtin_amdgcn_wmma_f32_16x16x32_bf16(
                    false, A.v, false, B1[k1][nt], (short)0, acc[nt], false, false);
        }
        // relu + store bf16 (C layout: elem i -> row 8*half+i, col l16)
#pragma unroll
        for (int nt = 0; nt < 2; ++nt)
#pragma unroll
            for (int i = 0; i < 8; ++i) {
                int row = mt * 16 + 8 * half + i;
                int col = w * 32 + nt * 16 + l16;
                float v = acc[nt][i];
                v = v > 0.f ? v : 0.f;
                *(__bf16*)(smem + SH_OFF + row * SH_ROW_BYTES + col * 2) = (__bf16)v;
            }
    }

    // ---- B2 weight tiles (B1 now dead): wave's 16 proj-cols, 16 k-tiles ----
    v16bf B2[16];
#pragma unroll
    for (int k2 = 0; k2 < 16; ++k2)
        B2[k2] = load_btile(w2t, w * 16 + l16, HCOLS, k2 * 32, half);
    const float bias2 = b2[w * 16 + l16];

    __syncthreads();

    // ---- GEMM2: proj = h @ W2 + b2 ; overwrite sH rows with proj (f32) ----
#pragma unroll 1
    for (int mt = 0; mt < 8; ++mt) {
        v8f acc2 = (v8f){bias2, bias2, bias2, bias2, bias2, bias2, bias2, bias2};
        const int rowb = (mt * 16 + l16) * SH_ROW_BYTES;
#pragma unroll
        for (int k2 = 0; k2 < 16; ++k2) {
            union { v16bf v; u32x4 q[2]; } A;
            A.q[0] = *(const u32x4*)(smem + SH_OFF + rowb + (k2 * 16 + 8 * half) * 4);
            A.q[1] = *(const u32x4*)(smem + SH_OFF + rowb + (k2 * 16 + 4 + 8 * half) * 4);
            acc2 = __builtin_amdgcn_wmma_f32_16x16x32_bf16(
                false, A.v, false, B2[k2], (short)0, acc2, false, false);
        }
        __syncthreads();   // all waves done reading these sH rows before overwrite
#pragma unroll
        for (int i = 0; i < 8; ++i) {
            int row = mt * 16 + 8 * half + i;
            int col = w * 16 + l16;
            *(float*)(smem + SH_OFF + row * SH_ROW_BYTES + col * 4) = acc2[i];
        }
    }
    __syncthreads();

    // ---- segmented stats: thread -> (row-half rh, column); sorted rows ----
    {
        const int col = tid & 255;
        const int rh  = tid >> 8;           // 0 or 1: rows [64rh, 64rh+64)
        int curSeg = -1;
        float s = 0.f, ss = 0.f;
        float mx = -__builtin_inff(), mn = __builtin_inff();
        for (int r = rh * 64; r < rh * 64 + 64; ++r) {
            int rg = r0 + r;
            if (rg >= N_ROWS) break;
            int seg = sBat[r];
            float v = *(const float*)(smem + SH_OFF + r * SH_ROW_BYTES + col * 4);
            if (seg != curSeg) {
                if (curSeg >= 0) {
                    int idx = curSeg * SCOLS + col;
                    atomicAdd(&sums[idx], s);
                    atomicAdd(&sumsq[idx], ss);
                    atomicMax(&maxb[idx], fmap(mx));
                    atomicMin(&minb[idx], fmap(mn));
                }
                curSeg = seg;
                s = 0.f; ss = 0.f;
                mx = -__builtin_inff(); mn = __builtin_inff();
            }
            s += v; ss += v * v;
            mx = fmaxf(mx, v); mn = fminf(mn, v);
        }
        if (curSeg >= 0) {
            int idx = curSeg * SCOLS + col;
            atomicAdd(&sums[idx], s);
            atomicAdd(&sumsq[idx], ss);
            atomicMax(&maxb[idx], fmap(mx));
            atomicMin(&minb[idx], fmap(mn));
        }
    }
}

// ------------------------------------------------------------------
// Kernel 2: finalize — counts via binary search on sorted batch,
//           mean/std, softmax(stat_weights), sketch
// ------------------------------------------------------------------
__device__ __forceinline__ int lower_bound_dev(const int* a, int n, int key) {
    int lo = 0, hi = n;
    while (lo < hi) {
        int mid = (lo + hi) >> 1;
        if (a[mid] < key) lo = mid + 1; else hi = mid;
    }
    return lo;
}

__global__ void finalize_kernel(const int* __restrict__ batch, const float* __restrict__ sw,
                                const float* __restrict__ sums, const float* __restrict__ sumsq,
                                const unsigned* __restrict__ maxb, const unsigned* __restrict__ minb,
                                float* __restrict__ out) {
    __shared__ float scount;
    const int seg = blockIdx.x;
    if (threadIdx.x == 0) {
        int a = lower_bound_dev(batch, N_ROWS, seg);
        int b = lower_bound_dev(batch, N_ROWS, seg + 1);
        scount = (float)(b - a);
    }
    __syncthreads();
    const float c = scount;
    const int col = threadIdx.x;
    const int idx = seg * SCOLS + col;

    float sum  = sums[idx];
    float sq   = sumsq[idx];
    float mean = sum / c;
    float var  = fmaxf(sq - c * mean * mean, 0.f) / (c - 1.f);
    float sd   = sqrtf(var);
    float mx   = funmap(maxb[idx]);
    float mn   = funmap(minb[idx]);

    float w0 = sw[0], w1 = sw[1], w2 = sw[2], w3 = sw[3];
    float m  = fmaxf(fmaxf(w0, w1), fmaxf(w2, w3));
    float e0 = __expf(w0 - m), e1 = __expf(w1 - m), e2 = __expf(w2 - m), e3 = __expf(w3 - m);
    float inv = 1.0f / (e0 + e1 + e2 + e3);

    out[idx] = (e0 * mean + e1 * mx + e2 * mn + e3 * sd) * inv;
}

// ------------------------------------------------------------------
extern "C" void kernel_launch(void* const* d_in, const int* in_sizes, int n_in,
                              void* d_out, int out_size, void* d_ws, size_t ws_size,
                              hipStream_t stream) {
    const float* x     = (const float*)d_in[0];
    const int*   batch = (const int*)d_in[1];
    // d_in[2] = num_graphs (fixed at 64)
    const float* W1 = (const float*)d_in[3];
    const float* b1 = (const float*)d_in[4];
    const float* W2 = (const float*)d_in[5];
    const float* b2 = (const float*)d_in[6];
    const float* sw = (const float*)d_in[7];

    unsigned char* ws = (unsigned char*)d_ws;
    __bf16*   w1t   = (__bf16*)(ws + WS_W1T);
    __bf16*   w2t   = (__bf16*)(ws + WS_W2T);
    float*    sums  = (float*)(ws + WS_SUMS);
    float*    sumsq = (float*)(ws + WS_SUMSQ);
    unsigned* maxb  = (unsigned*)(ws + WS_MAXB);
    unsigned* minb  = (unsigned*)(ws + WS_MINB);

    prep_kernel<<<256, 256, 0, stream>>>(W1, W2, w1t, w2t, sums, sumsq, maxb, minb);

    int nTiles = (N_ROWS + 127) / 128;   // 3907
    mlp_stats_kernel<<<nTiles, 512, LDS_BYTES, stream>>>(
        x, batch, w1t, b1, w2t, b2, sums, sumsq, maxb, minb);

    finalize_kernel<<<BSEG, SCOLS, 0, stream>>>(batch, sw, sums, sumsq, maxb, minb,
                                                (float*)d_out);
}
